// Encoder_78821239816499
// MI455X (gfx1250) — matrix operand order repaired and checked by reading, not compiled, not run on hardware
//
#include <hip/hip_runtime.h>
#include <math.h>

#define Dd  256
#define Hh  512
#define Bb  64
#define Tt  512
#define G4H 2048   // 4*H

typedef __attribute__((ext_vector_type(16))) __bf16 v16bf;
typedef __attribute__((ext_vector_type(8)))  __bf16 v8bf;
typedef __attribute__((ext_vector_type(8)))  float  v8f;

__device__ __forceinline__ float sigmoidf_(float x) {
  return 1.0f / (1.0f + __expf(-x));
}

// ---------------------------------------------------------------------------
// One LSTM time step (fused input+recurrent GEMM), all-bf16 operands:
//   gates[dir][b][n] = [x_t | h_prev] @ W^T + bias      (W is [4H, F+H] bf16)
// One wave computes one 16x16 tile via v_wmma_f32_16x16x32_bf16.
// A-operand (16x32 bf16), lane l: m = l&15, half = l>>4,
//   elems 0..7  <- K = kc + half*16 + j
//   elems 8..15 <- K = kc + half*16 + 8 + j
// B-operand (32x16 bf16), lane l: n = l&15, half = l>>4,
//   elems 0..15 <- K = kc + half*16 + j  (contiguous row of N-major weights)
// C/D (16x16 f32), lane l, vgpr r: m = half*8 + r, n = l&15.
// F is a template constant so the K loop fully unrolls.
// ---------------------------------------------------------------------------
template <int F>
__global__ __launch_bounds__(256)
void lstm_gates_k(const __bf16* __restrict__ x0p, const __bf16* __restrict__ x1p,
                  int t0, int t1,
                  const __bf16* __restrict__ h0s, const __bf16* __restrict__ h1s,
                  const __bf16* __restrict__ W0, const __bf16* __restrict__ W1,
                  const float* __restrict__ bias0, const float* __restrict__ bias1,
                  float* __restrict__ gates)
{
  constexpr int Ktot = F + Hh;

  const int dir      = blockIdx.y;
  const __bf16* xp   = dir ? x1p   : x0p;
  const int     t    = dir ? t1    : t0;
  const __bf16* hs   = dir ? h1s   : h0s;
  const __bf16* W    = dir ? W1    : W0;
  const float*  bias = dir ? bias1 : bias0;

  const int wid   = blockIdx.x * (blockDim.x >> 5) + (threadIdx.x >> 5);
  const int lane  = threadIdx.x & 31;
  const int NT    = G4H / 16;                 // 128 N-tiles
  const int mtile = wid / NT;                 // 0..3
  const int ntile = wid - mtile * NT;         // 0..127
  const int half  = lane >> 4;
  const int r     = lane & 15;

  const __bf16* xrow = xp + ((long)(mtile * 16 + r) * Tt + t) * F;
  const __bf16* hrow = hs + (mtile * 16 + r) * Hh;
  const __bf16* wrow = W  + (long)(ntile * 16 + r) * Ktot;

  v8f acc = {};
#pragma unroll
  for (int kc = 0; kc < Ktot; kc += 32) {
    // F and H are multiples of 32, so each K-chunk is purely x or purely h.
    const __bf16* arow = (kc < F) ? (xrow + kc) : (hrow + (kc - F));
    const v8bf lo = *(const v8bf*)(arow + half * 16);
    const v8bf hi = *(const v8bf*)(arow + half * 16 + 8);
    v16bf a;
#pragma unroll
    for (int j = 0; j < 8; ++j) { a[j] = lo[j]; a[8 + j] = hi[j]; }
    const v16bf bmat = *(const v16bf*)(wrow + kc + half * 16);
    acc = __builtin_amdgcn_wmma_f32_16x16x32_bf16(false, a, false, bmat,
                                                  (short)0, acc, false, false);
  }

  const int   n  = ntile * 16 + r;
  const float bn = bias[n];
  float* grow = gates + (long)dir * Bb * G4H;
#pragma unroll
  for (int rr = 0; rr < 8; ++rr) {
    const int mrow = mtile * 16 + half * 8 + rr;
    grow[(long)mrow * G4H + n] = acc[rr] + bn;
  }
}

// ---------------------------------------------------------------------------
// LSTM pointwise: gate nonlinearities, c/h update, emit h as bf16 state (for
// the next step's WMMA) and into the consumer buffer (bf16 or f32).
// Handles 1 or 2 directions depending on grid size.
// ---------------------------------------------------------------------------
__global__ __launch_bounds__(256)
void lstm_pointwise_k(const float* __restrict__ gates,
                      __bf16* __restrict__ hbf0, float* __restrict__ c0s,
                      __bf16* __restrict__ hbf1, float* __restrict__ c1s,
                      __bf16* __restrict__ out_bf,   // nullable
                      float*  __restrict__ out_f32,  // nullable
                      float*  __restrict__ hsave,    // nullable (f32 hT state)
                      long out_bstride, long out_off0, long out_off1)
{
  const int idx = blockIdx.x * blockDim.x + threadIdx.x;
  const int dir = idx / (Bb * Hh);
  const int rem = idx - dir * (Bb * Hh);
  const int b   = rem / Hh;
  const int j   = rem - b * Hh;

  const float* g = gates + ((long)dir * Bb + b) * G4H;
  __bf16* hbv = dir ? hbf1 : hbf0;
  float*  csv = dir ? c1s  : c0s;

  const float ig = sigmoidf_(g[j]);
  const float fg = sigmoidf_(g[Hh + j]);
  const float gg = tanhf(g[2 * Hh + j]);
  const float og = sigmoidf_(g[3 * Hh + j]);
  const float c  = fg * csv[b * Hh + j] + ig * gg;
  const float h  = og * tanhf(c);
  csv[b * Hh + j] = c;
  hbv[b * Hh + j] = (__bf16)h;

  const long o = (long)b * out_bstride + (dir ? out_off1 : out_off0) + j;
  if (out_bf)  out_bf[o]  = (__bf16)h;
  if (out_f32) out_f32[o] = h;
  if (hsave)   hsave[b * Hh + j] = h;
}

// ---------------------------------------------------------------------------
// Projection: proj[m][n] = tanh( bi[m][:] @ Wp[n][:] + bias[n] )  (bf16 out)
// M = B*T = 32768, N = H = 512, K = 2H = 1024.
// Register-blocked: one wave computes 4 adjacent 16x16 N-tiles, reusing A.
// ---------------------------------------------------------------------------
__global__ __launch_bounds__(256)
void proj_k(const __bf16* __restrict__ bi, const __bf16* __restrict__ Wp,
            const float* __restrict__ bias, __bf16* __restrict__ outp)
{
  constexpr int K  = 2 * Hh;
  constexpr int NG = (Hh / 16) / 4;          // 8 groups of 4 N-tiles

  const int wid   = blockIdx.x * (blockDim.x >> 5) + (threadIdx.x >> 5);
  const int lane  = threadIdx.x & 31;
  const int mtile = wid / NG;                // 0..2047
  const int ngrp  = wid - mtile * NG;        // 0..7
  const int half  = lane >> 4;
  const int r     = lane & 15;
  const int n0    = ngrp * 64 + r;           // first of 4 weight rows (stride 16)

  const __bf16* arowb = bi + (long)(mtile * 16 + r) * K;
  const __bf16* w0 = Wp + (long)(n0 +  0) * K;
  const __bf16* w1 = Wp + (long)(n0 + 16) * K;
  const __bf16* w2 = Wp + (long)(n0 + 32) * K;
  const __bf16* w3 = Wp + (long)(n0 + 48) * K;

  v8f acc0 = {}, acc1 = {}, acc2 = {}, acc3 = {};
#pragma unroll 4
  for (int kc = 0; kc < K; kc += 32) {
    const __bf16* arow = arowb + kc;
    const v8bf lo = *(const v8bf*)(arow + half * 16);
    const v8bf hi = *(const v8bf*)(arow + half * 16 + 8);
    v16bf a;
#pragma unroll
    for (int j = 0; j < 8; ++j) { a[j] = lo[j]; a[8 + j] = hi[j]; }
    const int ko = kc + half * 16;
    const v16bf b0 = *(const v16bf*)(w0 + ko);
    const v16bf b1 = *(const v16bf*)(w1 + ko);
    const v16bf b2 = *(const v16bf*)(w2 + ko);
    const v16bf b3 = *(const v16bf*)(w3 + ko);
    acc0 = __builtin_amdgcn_wmma_f32_16x16x32_bf16(false, a, false, b0, (short)0, acc0, false, false);
    acc1 = __builtin_amdgcn_wmma_f32_16x16x32_bf16(false, a, false, b1, (short)0, acc1, false, false);
    acc2 = __builtin_amdgcn_wmma_f32_16x16x32_bf16(false, a, false, b2, (short)0, acc2, false, false);
    acc3 = __builtin_amdgcn_wmma_f32_16x16x32_bf16(false, a, false, b3, (short)0, acc3, false, false);
  }

  const float bn0 = bias[n0];
  const float bn1 = bias[n0 + 16];
  const float bn2 = bias[n0 + 32];
  const float bn3 = bias[n0 + 48];
#pragma unroll
  for (int rr = 0; rr < 8; ++rr) {
    const long mrow = (long)(mtile * 16 + half * 8 + rr) * Hh;
    outp[mrow + n0 +  0] = (__bf16)tanhf(acc0[rr] + bn0);
    outp[mrow + n0 + 16] = (__bf16)tanhf(acc1[rr] + bn1);
    outp[mrow + n0 + 32] = (__bf16)tanhf(acc2[rr] + bn2);
    outp[mrow + n0 + 48] = (__bf16)tanhf(acc3[rr] + bn3);
  }
}

// ---------------------------------------------------------------------------
// Prep: concat [A|B] rows and convert f32 -> bf16 (cb==0 => plain convert).
// ---------------------------------------------------------------------------
__global__ void pack2_bf16_k(const float* __restrict__ A, int ca,
                             const float* __restrict__ Bm, int cb,
                             __bf16* __restrict__ out, long n)
{
  long i = (long)blockIdx.x * blockDim.x + threadIdx.x;
  if (i >= n) return;
  if (cb == 0) { out[i] = (__bf16)A[i]; return; }
  const int  K   = ca + cb;
  const long row = i / K;
  const int  k   = (int)(i - row * K);
  const float v  = (k < ca) ? A[row * (long)ca + k] : Bm[row * (long)cb + (k - ca)];
  out[i] = (__bf16)v;
}

__global__ void add_bias_k(const float* __restrict__ a, const float* __restrict__ b,
                           float* __restrict__ o, int n)
{
  int i = blockIdx.x * blockDim.x + threadIdx.x;
  if (i < n) o[i] = a[i] + b[i];
}

__global__ void zero_f_k(float* __restrict__ p, long n)
{
  long i = (long)blockIdx.x * blockDim.x + threadIdx.x;
  if (i < n) p[i] = 0.0f;
}

__global__ void zero_bf_k(__bf16* __restrict__ p, long n)
{
  long i = (long)blockIdx.x * blockDim.x + threadIdx.x;
  if (i < n) p[i] = (__bf16)0.0f;
}

__global__ void copy_k(const float* __restrict__ s, float* __restrict__ d, long n)
{
  long i = (long)blockIdx.x * blockDim.x + threadIdx.x;
  if (i < n) d[i] = s[i];
}

// ---------------------------------------------------------------------------
extern "C" void kernel_launch(void* const* d_in, const int* in_sizes, int n_in,
                              void* d_out, int out_size, void* d_ws, size_t ws_size,
                              hipStream_t stream)
{
  (void)in_sizes; (void)n_in; (void)out_size; (void)ws_size;

  const float* inputs = (const float*)d_in[0];
  const float* w_ih_f = (const float*)d_in[1];
  const float* w_hh_f = (const float*)d_in[2];
  const float* b_ih_f = (const float*)d_in[3];
  const float* b_hh_f = (const float*)d_in[4];
  const float* w_ih_b = (const float*)d_in[5];
  const float* w_hh_b = (const float*)d_in[6];
  const float* b_ih_b = (const float*)d_in[7];
  const float* b_hh_b = (const float*)d_in[8];
  const float* w_proj = (const float*)d_in[9];
  const float* b_proj = (const float*)d_in[10];
  const float* w_ih_c = (const float*)d_in[11];
  const float* w_hh_c = (const float*)d_in[12];
  const float* b_ih_c = (const float*)d_in[13];
  const float* b_hh_c = (const float*)d_in[14];

  // -------- workspace carve (256B aligned) --------
  char* p = (char*)d_ws;
  auto carve = [&p](size_t bytes) -> void* {
    void* r = (void*)p;
    p += (bytes + 255) & ~(size_t)255;
    return r;
  };

  __bf16* Wf = (__bf16*)carve((size_t)G4H * (Dd + Hh) * sizeof(__bf16));
  __bf16* Wb = (__bf16*)carve((size_t)G4H * (Dd + Hh) * sizeof(__bf16));
  __bf16* Wc = (__bf16*)carve((size_t)G4H * (2 * Hh)  * sizeof(__bf16));
  __bf16* Wp = (__bf16*)carve((size_t)Hh  * (2 * Hh)  * sizeof(__bf16));
  float* bias_f = (float*)carve((size_t)G4H * sizeof(float));
  float* bias_b = (float*)carve((size_t)G4H * sizeof(float));
  float* bias_c = (float*)carve((size_t)G4H * sizeof(float));
  // f32 states: cF, cB, hCf, cC  (hCf & cC adjacent -> single tail copy)
  float* statesF = (float*)carve((size_t)4 * Bb * Hh * sizeof(float));
  float* cF  = statesF + 0 * Bb * Hh;
  float* cB  = statesF + 1 * Bb * Hh;
  float* hCf = statesF + 2 * Bb * Hh;
  float* cC  = statesF + 3 * Bb * Hh;
  // bf16 h states (WMMA A operands)
  __bf16* statesB = (__bf16*)carve((size_t)3 * Bb * Hh * sizeof(__bf16));
  __bf16* hFb = statesB + 0 * Bb * Hh;
  __bf16* hBb = statesB + 1 * Bb * Hh;
  __bf16* hCb = statesB + 2 * Bb * Hh;
  float*  gates = (float*)carve((size_t)2 * Bb * G4H * sizeof(float));
  __bf16* xbf   = (__bf16*)carve((size_t)Bb * Tt * Dd * sizeof(__bf16));
  __bf16* bi    = (__bf16*)carve((size_t)Bb * Tt * 2 * Hh * sizeof(__bf16));
  __bf16* proj  = (__bf16*)carve((size_t)Bb * Tt * Hh * sizeof(__bf16));

  float* out_enc = (float*)d_out;
  float* out_hc  = out_enc + (size_t)Bb * Tt * Hh;   // hT then cT

  // -------- prep: bf16 packing of weights/inputs, bias sums, state zeroing --
  {
    long n = (long)G4H * (Dd + Hh);
    pack2_bf16_k<<<(int)((n + 255) / 256), 256, 0, stream>>>(w_ih_f, Dd, w_hh_f, Hh, Wf, n);
    pack2_bf16_k<<<(int)((n + 255) / 256), 256, 0, stream>>>(w_ih_b, Dd, w_hh_b, Hh, Wb, n);
    n = (long)G4H * (2 * Hh);
    pack2_bf16_k<<<(int)((n + 255) / 256), 256, 0, stream>>>(w_ih_c, Hh, w_hh_c, Hh, Wc, n);
    n = (long)Hh * (2 * Hh);
    pack2_bf16_k<<<(int)((n + 255) / 256), 256, 0, stream>>>(w_proj, 2 * Hh, nullptr, 0, Wp, n);
    n = (long)Bb * Tt * Dd;
    pack2_bf16_k<<<(int)((n + 255) / 256), 256, 0, stream>>>(inputs, Dd, nullptr, 0, xbf, n);
    add_bias_k<<<(G4H + 255) / 256, 256, 0, stream>>>(b_ih_f, b_hh_f, bias_f, G4H);
    add_bias_k<<<(G4H + 255) / 256, 256, 0, stream>>>(b_ih_b, b_hh_b, bias_b, G4H);
    add_bias_k<<<(G4H + 255) / 256, 256, 0, stream>>>(b_ih_c, b_hh_c, bias_c, G4H);
    zero_f_k<<<(4 * Bb * Hh + 255) / 256, 256, 0, stream>>>(statesF, (long)4 * Bb * Hh);
    zero_bf_k<<<(3 * Bb * Hh + 255) / 256, 256, 0, stream>>>(statesB, (long)3 * Bb * Hh);
  }

  const int gatesGrid = (Bb / 16) * (G4H / 16) / 8;   // 512 waves, 8 per block

  // -------- BiLSTM: fwd (dir 0) and time-reversed bwd (dir 1) per step --------
  for (int s = 0; s < Tt; ++s) {
    const int tf = s, tb = Tt - 1 - s;
    lstm_gates_k<Dd><<<dim3(gatesGrid, 2), 256, 0, stream>>>(
        xbf, xbf, tf, tb, hFb, hBb, Wf, Wb, bias_f, bias_b, gates);
    // fwd h -> bi[b][tf][0:H];  bwd h -> bi[b][tb][H:2H] (undoes the reversal)
    lstm_pointwise_k<<<(2 * Bb * Hh) / 256, 256, 0, stream>>>(
        gates, hFb, cF, hBb, cB, bi, nullptr, nullptr,
        (long)Tt * 2 * Hh, (long)tf * 2 * Hh, (long)tb * 2 * Hh + Hh);
  }

  // -------- projection + tanh: proj = tanh(bi @ Wp^T + b_proj) --------
  {
    const long waves = (long)(Bb * Tt / 16) * ((Hh / 16) / 4);   // 16384
    proj_k<<<(int)(waves / 8), 256, 0, stream>>>(bi, Wp, b_proj, proj);
  }

  // -------- LSTMCell scan over proj, h written straight to d_out --------
  for (int s = 0; s < Tt; ++s) {
    lstm_gates_k<Hh><<<dim3(gatesGrid, 1), 256, 0, stream>>>(
        proj, proj, s, s, hCb, hCb, Wc, Wc, bias_c, bias_c, gates);
    lstm_pointwise_k<<<(Bb * Hh) / 256, 256, 0, stream>>>(
        gates, hCb, cC, hCb, cC, nullptr, out_enc, hCf,
        (long)Tt * Hh, (long)s * Hh, 0);
  }

  // -------- (hT, cT) tail --------
  copy_k<<<(2 * Bb * Hh + 255) / 256, 256, 0, stream>>>(hCf, out_hc, (long)2 * Bb * Hh);
}
